// TensorTrainLMN_67568425501235
// MI455X (gfx1250) — compile-verified, matrix operand order checked by reading, not run on hardware
//
#include <hip/hip_runtime.h>
#include <hip/hip_bf16.h>

// Problem constants (fixed by the reference)
#define BS     32768
#define N_CH   16
#define HSZ    128
#define RK     64
#define N_AGGR 4
#define TSZ    128
#define KTOT   2176          // 16*128 (children) + 128 (type)
#define NOUT   512           // 4 aggr * 128
#define KSTEPS 68            // 2176 / 32
#define LDSTR  40            // LDS row stride in bf16 (32 + 8 pad -> conflict-free b128 reads)

typedef __attribute__((ext_vector_type(16))) __bf16 v16bf;
typedef __attribute__((ext_vector_type(8)))  float  v8f;

#if __has_builtin(__builtin_amdgcn_tensor_load_to_lds)
#define USE_TDM 1
#else
#define USE_TDM 0
#endif

// ---------- fp32 -> bf16 ----------
__device__ __forceinline__ unsigned bfbits(float f) {
    union { float f; unsigned u; } a; a.f = f;
    unsigned r = a.u + 0x7FFFu + ((a.u >> 16) & 1u);
    return r >> 16;
}
#if __has_builtin(__builtin_amdgcn_perm)
// (bf16(y) << 16) | bf16(x) in 3 VALU ops: two rounding adds + one v_perm_b32.
__device__ __forceinline__ unsigned pk2(float x, float y) {
    union { float f; unsigned u; } ax, ay; ax.f = x; ay.f = y;
    return __builtin_amdgcn_perm(ay.u + 0x8000u, ax.u + 0x8000u, 0x07060302u);
}
#elif __has_builtin(__builtin_amdgcn_cvt_pk_bf16_f32)
__device__ __forceinline__ unsigned pk2(float x, float y) {
    typedef __attribute__((ext_vector_type(2))) __bf16 v2bf;
    union { v2bf v; unsigned u; } c;
    c.v = __builtin_amdgcn_cvt_pk_bf16_f32(x, y);
    return c.u;
}
#else
__device__ __forceinline__ unsigned pk2(float x, float y) {
    return bfbits(x) | (bfbits(y) << 16);
}
#endif
__device__ __forceinline__ __bf16 f2bf(float f) {
    union { unsigned short s; __bf16 b; } o;
    o.s = (unsigned short)bfbits(f);
    return o.b;
}

// ================================================================
// S1: B-power chain.  Bpow[a][j] = B[a]^j, j = 0..16  (fp32, 64x64)
// ================================================================
__global__ void tt_s1_bpow(const float* __restrict__ B, float* __restrict__ Bpow) {
    const int a = blockIdx.x, tid = threadIdx.x;
    __shared__ float Bl[RK * RK];
    __shared__ float Pl[RK * RK];
    float* out0 = Bpow + (size_t)a * 17 * RK * RK;
    for (int idx = tid; idx < RK * RK; idx += 256) {
        Bl[idx] = B[(size_t)a * RK * RK + idx];
        float id = ((idx >> 6) == (idx & 63)) ? 1.f : 0.f;
        Pl[idx] = id;
        out0[idx] = id;                         // B^0 = I
    }
    __syncthreads();
    for (int j = 1; j <= 16; ++j) {
        float tmp[16];
        #pragma unroll
        for (int t = 0; t < 16; ++t) {
            int idx = tid + t * 256;
            int i = idx >> 6, r = idx & 63;
            float s = 0.f;
            #pragma unroll 8
            for (int q = 0; q < RK; ++q) s += Pl[i * RK + q] * Bl[q * RK + r];
            tmp[t] = s;
        }
        __syncthreads();
        float* oj = out0 + (size_t)j * RK * RK;
        #pragma unroll
        for (int t = 0; t < 16; ++t) {
            int idx = tid + t * 256;
            Pl[idx] = tmp[t];
            oj[idx] = tmp[t];
        }
        __syncthreads();
    }
}

// ================================================================
// S2: folded weights.  chunk c<16: W_blk = A[a] @ B^(15-c) @ U_out[a]
//     chunk 16:        W_blk = U_type[a] @ B^16 @ U_out[a]
// Stored bf16, layout Wt[n][K]  (n = a*128+h, K = chunk*128+i)
// ================================================================
__global__ void tt_s2_weights(const float* __restrict__ Aw,
                              const float* __restrict__ U_type,
                              const float* __restrict__ U_output,
                              const float* __restrict__ Bpow,
                              __bf16* __restrict__ Wt) {
    const int bb = blockIdx.x, tid = threadIdx.x;
    const int a = bb / 17, chunk = bb - a * 17;
    const int p = (chunk < 16) ? (15 - chunk) : 16;
    const float* Msrc = (chunk < 16) ? (Aw + (size_t)a * HSZ * RK)
                                     : (U_type + (size_t)a * TSZ * RK);
    __shared__ float Pl[RK * RK];
    __shared__ float T1[HSZ * RK];
    __shared__ float Uo[RK * HSZ];
    const float* Pg = Bpow + ((size_t)a * 17 + p) * RK * RK;
    for (int idx = tid; idx < RK * RK; idx += 256) Pl[idx] = Pg[idx];
    for (int idx = tid; idx < RK * HSZ; idx += 256)
        Uo[idx] = U_output[(size_t)a * RK * HSZ + idx];
    __syncthreads();
    for (int idx = tid; idx < HSZ * RK; idx += 256) {
        int i = idx >> 6, r = idx & 63;
        float s = 0.f;
        #pragma unroll 8
        for (int q = 0; q < RK; ++q) s += Msrc[i * RK + q] * Pl[q * RK + r];
        T1[idx] = s;
    }
    __syncthreads();
    for (int idx = tid; idx < HSZ * HSZ; idx += 256) {
        int i = idx >> 7, h = idx & 127;
        float s = 0.f;
        #pragma unroll 8
        for (int r = 0; r < RK; ++r) s += T1[i * RK + r] * Uo[r * HSZ + h];
        Wt[(size_t)(a * HSZ + h) * KTOT + chunk * HSZ + i] = f2bf(s);
    }
}

// ================================================================
// S3: bias[n] = ((A_b @ sum_{j<16} B^j + b_type @ B^16) @ U_out + b_out)
// ================================================================
__global__ void tt_s3_bias(const float* __restrict__ Bpow,
                           const float* __restrict__ A_b,
                           const float* __restrict__ b_type,
                           const float* __restrict__ U_output,
                           const float* __restrict__ b_output,
                           float* __restrict__ bias) {
    const int a = blockIdx.x, tid = threadIdx.x;
    __shared__ float S[RK * RK];
    __shared__ float v[RK];
    const float* base = Bpow + (size_t)a * 17 * RK * RK;
    for (int idx = tid; idx < RK * RK; idx += 128) {
        float s = 0.f;
        #pragma unroll
        for (int j = 0; j < 16; ++j) s += base[(size_t)j * RK * RK + idx];
        S[idx] = s;
    }
    __syncthreads();
    if (tid < RK) {
        const float* B16 = base + (size_t)16 * RK * RK;
        float acc = 0.f;
        #pragma unroll 8
        for (int q = 0; q < RK; ++q)
            acc += A_b[a * RK + q] * S[q * RK + tid] +
                   b_type[a * RK + q] * B16[q * RK + tid];
        v[tid] = acc;
    }
    __syncthreads();
    {
        int h = tid;
        float acc = b_output[a * HSZ + h];
        #pragma unroll 8
        for (int r = 0; r < RK; ++r)
            acc += v[r] * U_output[((size_t)a * RK + r) * HSZ + h];
        bias[a * HSZ + h] = acc;
    }
}

// ================================================================
// Main GEMM:  out(32768 x 512) = X(32768 x 2176) @ W(2176 x 512) + bias
// ================================================================

#if USE_TDM
typedef __attribute__((ext_vector_type(4))) unsigned int u32x4;
typedef __attribute__((ext_vector_type(8))) int i32x8;
typedef __attribute__((ext_vector_type(4))) int i32x4;

// One TDM descriptor: 2-D tile 32(K) x 128(n) bf16 out of Wt (row stride KTOT),
// DMA'd into LDS with padding: 16 DWORDs data + 4 DWORDs pad -> 40-bf16 rows.
__device__ __forceinline__ void tdm_load_b(const __bf16* src, void* lds_dst) {
    const unsigned lds = (unsigned)(size_t)lds_dst;      // low 32 bits = LDS byte addr
    const unsigned long long ga = (unsigned long long)(size_t)src;
    u32x4 g0;
    g0[0] = 1u;                                          // count=1, user descriptor
    g0[1] = lds;                                         // lds_addr [63:32]
    g0[2] = (unsigned)ga;                                // global_addr [95:64]
    g0[3] = (unsigned)((ga >> 32) & 0x01FFFFFFu) | (2u << 30);  // addr hi | type=2
    i32x8 g1;
    g1[0] = (1 << 16)          // data_size = 1 (2 bytes)
          | (1 << 20)          // pad_enable
          | (3 << 22)          // pad_interval: 16 DWORDs (64B) between pads
          | (3 << 25);         // pad_amount: 4 DWORDs (16B)
    g1[1] = (int)((KTOT & 0xFFFF) << 16);                          // tensor_dim0 lo16
    g1[2] = (int)((KTOT >> 16) | ((NOUT & 0xFFFF) << 16));         // dim0 hi | dim1 lo
    g1[3] = (int)((NOUT >> 16) | (32 << 16));                      // dim1 hi | tile_dim0=32
    g1[4] = 128;                                                   // tile_dim1=128, tile_dim2=0
    g1[5] = KTOT;                                                  // tensor_dim0_stride lo32
    g1[6] = 0;
    g1[7] = 0;
    i32x4 z4 = {0, 0, 0, 0};
#if __clang_major__ >= 23
    i32x8 z8 = {0, 0, 0, 0, 0, 0, 0, 0};
    __builtin_amdgcn_tensor_load_to_lds(g0, g1, z4, z4, z8, 0);
#else
    __builtin_amdgcn_tensor_load_to_lds(g0, g1, z4, z4, 0);
#endif
}
#endif // USE_TDM

// Fragment gather per documented wave32 16-bit layout:
// lanes 0-15 hold K 0-7 & 16-23, lanes 16-31 hold K 8-15 & 24-31.
__device__ __forceinline__ v16bf ldfrag(const __bf16* p, int row, int half) {
    union { uint4 q[2]; v16bf v; } u;
    const uint4* q = (const uint4*)(p + row * LDSTR);
    u.q[0] = q[half];
    u.q[1] = q[2 + half];
    return u.v;
}

__device__ __forceinline__ void load_a_tile(const float* __restrict__ nh,
                                            const float* __restrict__ te,
                                            int Mbase, int kk, int tid,
                                            float4* Areg) {
    if (kk < 64) {
        const float* s = nh + (size_t)Mbase * 2048 + (size_t)kk * 32;
        #pragma unroll
        for (int t = 0; t < 4; ++t) {
            int idx = tid + t * 256, row = idx >> 3, c4 = idx & 7;
            Areg[t] = *(const float4*)(s + (size_t)row * 2048 + c4 * 4);
        }
    } else {
        const float* s = te + (size_t)Mbase * 128 + (size_t)(kk - 64) * 32;
        #pragma unroll
        for (int t = 0; t < 4; ++t) {
            int idx = tid + t * 256, row = idx >> 3, c4 = idx & 7;
            Areg[t] = *(const float4*)(s + (size_t)row * 128 + c4 * 4);
        }
    }
}

__device__ __forceinline__ void store_a_tile(__bf16* dst, int tid, const float4* Areg) {
    #pragma unroll
    for (int t = 0; t < 4; ++t) {
        int idx = tid + t * 256, row = idx >> 3, c4 = idx & 7;
        uint2 u;
        u.x = pk2(Areg[t].x, Areg[t].y);
        u.y = pk2(Areg[t].z, Areg[t].w);
        *(uint2*)&dst[row * LDSTR + c4 * 4] = u;
    }
}

#if !USE_TDM
__device__ __forceinline__ void load_b_tile(const __bf16* __restrict__ Wt,
                                            int Nbase, int kk, int tid,
                                            uint4* Breg) {
    const __bf16* s = Wt + (size_t)Nbase * KTOT + (size_t)kk * 32;
    #pragma unroll
    for (int t = 0; t < 2; ++t) {
        int idx = tid + t * 256, n = idx >> 2, kc = idx & 3;
        Breg[t] = *(const uint4*)(s + (size_t)n * KTOT + kc * 8);
    }
}
__device__ __forceinline__ void store_b_tile(__bf16* dst, int tid, const uint4* Breg) {
    #pragma unroll
    for (int t = 0; t < 2; ++t) {
        int idx = tid + t * 256, n = idx >> 2, kc = idx & 3;
        *(uint4*)&dst[n * LDSTR + kc * 8] = Breg[t];
    }
}
#endif

__global__ __launch_bounds__(256)
void tt_main_gemm(const float* __restrict__ nh, const float* __restrict__ te,
                  const __bf16* __restrict__ Wt, const float* __restrict__ bias,
                  float* __restrict__ out) {
    const int tid  = threadIdx.x;
    const int lane = tid & 31, wave = tid >> 5;
    const int half = lane >> 4, lrow = lane & 15;
    const int wm = wave & 3, wn = wave >> 2;           // 4 waves on M, 2 on N
    const int Mbase = (blockIdx.x >> 2) * 128;
    const int Nbase = (blockIdx.x & 3) * 128;

    __shared__ __bf16 Asm[2][128 * LDSTR];
    __shared__ __bf16 Bsm[2][128 * LDSTR];

    v8f acc[2][4];
    #pragma unroll
    for (int mt = 0; mt < 2; ++mt)
        #pragma unroll
        for (int nt = 0; nt < 4; ++nt) acc[mt][nt] = (v8f)0.0f;

    float bv[4];
    #pragma unroll
    for (int nt = 0; nt < 4; ++nt)
        bv[nt] = bias[Nbase + wn * 64 + nt * 16 + lrow];

    // ---- prologue: stage K-step 0 ----
    {
#if USE_TDM
        if (wave == 0)
            tdm_load_b(Wt + (size_t)Nbase * KTOT, &Bsm[0][0]);
#else
        uint4 Breg[2];
        load_b_tile(Wt, Nbase, 0, tid, Breg);
        store_b_tile(&Bsm[0][0], tid, Breg);
#endif
        float4 Areg[4];
        load_a_tile(nh, te, Mbase, 0, tid, Areg);
        store_a_tile(&Asm[0][0], tid, Areg);
#if USE_TDM
        __builtin_amdgcn_s_wait_tensorcnt(0);
#endif
    }
    __syncthreads();

    for (int kk = 0; kk < KSTEPS; ++kk) {
        const int cur = kk & 1, nxt = cur ^ 1;
        float4 Areg[4];
#if !USE_TDM
        uint4 Breg[2];
#endif
        if (kk + 1 < KSTEPS) {
#if USE_TDM
            if (wave == 0)      // DMA next B tile; overlaps the WMMAs below
                tdm_load_b(Wt + (size_t)Nbase * KTOT + (size_t)(kk + 1) * 32,
                           &Bsm[nxt][0]);
#else
            load_b_tile(Wt, Nbase, kk + 1, tid, Breg);
#endif
            load_a_tile(nh, te, Mbase, kk + 1, tid, Areg);
        }
        if (kk + 3 < 64) {      // pull a future A tile toward L2
            const float* pf = nh + (size_t)Mbase * 2048 + (size_t)(kk + 3) * 32
                              + (size_t)(tid >> 3) * 2048 + (tid & 7) * 4;
            __builtin_prefetch(pf, 0, 1);
        }

        // fragments + 8 WMMAs
        v16bf af[2], bfr[4];
        #pragma unroll
        for (int mt = 0; mt < 2; ++mt)
            af[mt] = ldfrag(&Asm[cur][0], wm * 32 + mt * 16 + lrow, half);
        #pragma unroll
        for (int nt = 0; nt < 4; ++nt)
            bfr[nt] = ldfrag(&Bsm[cur][0], wn * 64 + nt * 16 + lrow, half);
        #pragma unroll
        for (int mt = 0; mt < 2; ++mt)
            #pragma unroll
            for (int nt = 0; nt < 4; ++nt)
                acc[mt][nt] = __builtin_amdgcn_wmma_f32_16x16x32_bf16(
                    false, af[mt], false, bfr[nt], (short)0, acc[mt][nt],
                    false, false);

        if (kk + 1 < KSTEPS) {
            __syncthreads();
            store_a_tile(&Asm[nxt][0], tid, Areg);
#if !USE_TDM
            store_b_tile(&Bsm[nxt][0], tid, Breg);
#else
            __builtin_amdgcn_s_wait_tensorcnt(0);
#endif
            __syncthreads();
        }
    }

    // epilogue: C layout — VGPR r: lanes 0-15 (M=r, N=lane), lanes 16-31 (M=8+r)
    #pragma unroll
    for (int mt = 0; mt < 2; ++mt) {
        const size_t rowb = (size_t)(Mbase + wm * 32 + mt * 16 + half * 8);
        #pragma unroll
        for (int nt = 0; nt < 4; ++nt) {
            const int col = Nbase + wn * 64 + nt * 16 + lrow;
            v8f a = acc[mt][nt];
            #pragma unroll
            for (int r = 0; r < 8; ++r)
                out[(rowb + r) * NOUT + col] = a[r] + bv[nt];
        }
    }
}

// ================================================================
extern "C" void kernel_launch(void* const* d_in, const int* in_sizes, int n_in,
                              void* d_out, int out_size, void* d_ws, size_t ws_size,
                              hipStream_t stream) {
    const float* nh       = (const float*)d_in[0];   // (BS, 16, 128)
    const float* te       = (const float*)d_in[1];   // (BS, 128)
    const float* Aw       = (const float*)d_in[2];   // (4, 128, 64)
    const float* Bw       = (const float*)d_in[3];   // (4, 64, 64)
    const float* A_b      = (const float*)d_in[4];   // (4, 1, 64)
    const float* U_type   = (const float*)d_in[5];   // (4, 128, 64)
    const float* b_type   = (const float*)d_in[6];   // (4, 1, 64)
    const float* U_output = (const float*)d_in[7];   // (4, 64, 128)
    const float* b_output = (const float*)d_in[8];   // (4, 1, 128)
    float* out = (float*)d_out;

    float*  Bpow = (float*)d_ws;                 // 4*17*4096 f32
    float*  bias = Bpow + 4 * 17 * 64 * 64;      // 512 f32
    __bf16* Wt   = (__bf16*)(bias + 512);        // 512*2176 bf16 (16B aligned)

    tt_s1_bpow   <<<4,    256, 0, stream>>>(Bw, Bpow);
    tt_s2_weights<<<68,   256, 0, stream>>>(Aw, U_type, U_output, Bpow, Wt);
    tt_s3_bias   <<<4,    128, 0, stream>>>(Bpow, A_b, b_type, U_output, b_output, bias);
    tt_main_gemm <<<1024, 256, 0, stream>>>(nh, te, Wt, bias, out);
    (void)in_sizes; (void)n_in; (void)out_size; (void)ws_size;
}
